// SlotAttention_64828236366131
// MI455X (gfx1250) — matrix operand order repaired and checked by reading, not compiled, not run on hardware
//
#include <hip/hip_runtime.h>

// Problem constants (from reference)
#define B_ 64
#define N_ 2048
#define D_ 256
#define S_ 8
#define ITERS_ 3
#define SCALE_ 0.0625f   // 256^-0.5

typedef __attribute__((ext_vector_type(16))) _Float16 v16h;
typedef __attribute__((ext_vector_type(8)))  _Float16 v8h;
typedef __attribute__((ext_vector_type(8)))  float    v8f;

// ---------------------------------------------------------------------------
// WMMA helpers: CDNA5 V_WMMA_F32_16X16X32_F16, wave32.
// A (16x32 f16): lane m=L&15; halves 0..15 = K kb..kb+15,
//                kb = (L&16)?16:0.
// B (32x16 f16): lane n=L&15; halves 0..15 = K kb..kb+15, kb = (L&16)?16:0.
// C/D (16x16 f32): lane col = L&15; vgpr r -> row r + ((L>>4)*8).
// ---------------------------------------------------------------------------
static __device__ inline v8f wmma16(v16h a, v16h b, v8f c) {
  return __builtin_amdgcn_wmma_f32_16x16x32_f16(false, a, false, b, (short)0, c,
                                                false, false);
}

static __device__ inline v16h load_a16(const _Float16* A, int lda, int k0, int lane) {
  int m  = lane & 15;
  int kb = (lane & 16) ? 16 : 0;
  const _Float16* p = A + (size_t)m * lda + k0 + kb;
  v8h lo = *(const v8h*)p;
  v8h hi = *(const v8h*)(p + 8);
  v16h r;
#pragma unroll
  for (int i = 0; i < 8; ++i) { r[i] = lo[i]; r[i + 8] = hi[i]; }
  return r;
}

static __device__ inline v16h load_b16(const _Float16* W, int ldw, int colbase,
                                       int k0, int lane) {
  int n  = lane & 15;
  int kb = (lane & 16) ? 16 : 0;
  return *(const v16h*)(W + (size_t)(colbase + n) * ldw + k0 + kb);
}

// ---------------------------------------------------------------------------
// Utility kernels
// ---------------------------------------------------------------------------
__global__ void zero_f4_k(float4* p, size_t n4) {
  size_t i = (size_t)blockIdx.x * blockDim.x + threadIdx.x;
  if (i < n4) p[i] = make_float4(0.f, 0.f, 0.f, 0.f);
}

__global__ void cvt_h_k(const float* __restrict__ s, _Float16* __restrict__ d, int n) {
  int i = blockIdx.x * blockDim.x + threadIdx.x;
  if (i < n) d[i] = (_Float16)s[i];
}

// slots = mu + softplus(sigma) * noise, into padded [B][16][D] (rows 8..15 stay 0)
__global__ void slot_init_k(const float* __restrict__ mu, const float* __restrict__ sigma,
                            const float* __restrict__ noise, float* __restrict__ slots) {
  int i = blockIdx.x * blockDim.x + threadIdx.x;  // over B*8*D
  if (i >= B_ * S_ * D_) return;
  int c = i & (D_ - 1);
  int s = (i >> 8) & 7;
  int b = i >> 11;
  float sg = sigma[c];
  float sp = (sg > 20.f) ? sg : log1pf(expf(sg));
  slots[((size_t)b * 16 + s) * D_ + c] = mu[c] + sp * noise[i];
}

__global__ void copy_out_k(const float* __restrict__ slots, float* __restrict__ out) {
  int i = blockIdx.x * blockDim.x + threadIdx.x;  // over B*8*D
  if (i >= B_ * S_ * D_) return;
  int c = i & (D_ - 1);
  int s = (i >> 8) & 7;
  int b = i >> 11;
  out[i] = slots[((size_t)b * 16 + s) * D_ + c];
}

// ---------------------------------------------------------------------------
// Fused LN(x) + K/V projection.  One block = one 16-row strip of [B*N, D].
// K stored row-major f16 [B*N][D]; V stored transposed f16 [B][D][N].
// ---------------------------------------------------------------------------
__global__ __launch_bounds__(256) void lnx_kv_k(
    const float* __restrict__ x, const float* __restrict__ g, const float* __restrict__ be,
    const _Float16* __restrict__ wk, const float* __restrict__ bk,
    const _Float16* __restrict__ wv, const float* __restrict__ bv,
    _Float16* __restrict__ kout, _Float16* __restrict__ vTout) {
  __shared__ _Float16 xn[16][264];  // padded stride vs bank conflicts
  int wave = threadIdx.x >> 5, lane = threadIdx.x & 31;
  size_t r0 = (size_t)blockIdx.x * 16;

  // LayerNorm: each wave handles 2 rows
#pragma unroll
  for (int j = 0; j < 2; ++j) {
    int lr = wave * 2 + j;
    const float* row = x + (r0 + lr) * D_;
    float v[8];
    float sum = 0.f, sq = 0.f;
#pragma unroll
    for (int i = 0; i < 8; ++i) {
      v[i] = row[lane * 8 + i];
      sum += v[i];
      sq += v[i] * v[i];
    }
#pragma unroll
    for (int off = 16; off; off >>= 1) {
      sum += __shfl_xor(sum, off, 32);
      sq  += __shfl_xor(sq, off, 32);
    }
    float mean = sum * (1.f / D_);
    float var  = sq * (1.f / D_) - mean * mean;
    float rstd = rsqrtf(var + 1e-5f);
#pragma unroll
    for (int i = 0; i < 8; ++i) {
      int c = lane * 8 + i;
      xn[lr][c] = (_Float16)((v[i] - mean) * rstd * g[c] + be[c]);
    }
  }
  __syncthreads();

  // GEMM: each wave does 2 column tiles for both K and V
  v8f ak0 = {}, ak1 = {}, av0 = {}, av1 = {};
  const _Float16* xs = &xn[0][0];
#pragma unroll
  for (int ks = 0; ks < 8; ++ks) {
    int k0 = ks * 32;
    v16h a   = load_a16(xs, 264, k0, lane);
    v16h bk0 = load_b16(wk, D_, wave * 32, k0, lane);
    v16h bk1 = load_b16(wk, D_, wave * 32 + 16, k0, lane);
    v16h bv0 = load_b16(wv, D_, wave * 32, k0, lane);
    v16h bv1 = load_b16(wv, D_, wave * 32 + 16, k0, lane);
    ak0 = wmma16(a, bk0, ak0);
    ak1 = wmma16(a, bk1, ak1);
    av0 = wmma16(a, bv0, av0);
    av1 = wmma16(a, bv1, av1);
  }

  int n16 = lane & 15, hi = lane >> 4;
  size_t b = r0 >> 11;            // row / N_
  int nbase = (int)(r0 & (N_ - 1));
  int c0 = wave * 32 + n16, c1 = wave * 32 + 16 + n16;
#pragma unroll
  for (int r = 0; r < 8; ++r) {
    int row = hi * 8 + r;
    size_t grow = r0 + row;
    kout[grow * D_ + c0] = (_Float16)(ak0[r] + bk[c0]);
    kout[grow * D_ + c1] = (_Float16)(ak1[r] + bk[c1]);
    vTout[(b * D_ + c0) * N_ + nbase + row] = (_Float16)(av0[r] + bv[c0]);
    vTout[(b * D_ + c1) * N_ + nbase + row] = (_Float16)(av1[r] + bv[c1]);
  }
}

// ---------------------------------------------------------------------------
// Row LayerNorm of padded [B][16][D] f32 (rows 0..7 valid) -> f16 LN output,
// plus optional plain f16 copy.  One wave per row.
// ---------------------------------------------------------------------------
__global__ __launch_bounds__(256) void ln16_k(
    const float* __restrict__ in, const float* __restrict__ g, const float* __restrict__ be,
    _Float16* __restrict__ out_ln, _Float16* __restrict__ out_plain) {
  int gw = blockIdx.x * 8 + (threadIdx.x >> 5);  // 0..511
  int lane = threadIdx.x & 31;
  int b = gw >> 3, s = gw & 7;
  size_t base = ((size_t)b * 16 + s) * D_;
  const float* row = in + base;
  float v[8];
  float sum = 0.f, sq = 0.f;
#pragma unroll
  for (int i = 0; i < 8; ++i) {
    v[i] = row[lane * 8 + i];
    sum += v[i];
    sq += v[i] * v[i];
  }
#pragma unroll
  for (int off = 16; off; off >>= 1) {
    sum += __shfl_xor(sum, off, 32);
    sq  += __shfl_xor(sq, off, 32);
  }
  float mean = sum * (1.f / D_);
  float var  = sq * (1.f / D_) - mean * mean;
  float rstd = rsqrtf(var + 1e-5f);
#pragma unroll
  for (int i = 0; i < 8; ++i) {
    int c = lane * 8 + i;
    out_ln[base + c] = (_Float16)((v[i] - mean) * rstd * g[c] + be[c]);
    if (out_plain) out_plain[base + c] = (_Float16)v[i];
  }
}

// ---------------------------------------------------------------------------
// Generic small GEMM: Out[b][row][col] = act( (A[b] @ W[b]^T)[row][col] + bias[col] )
// A: [b][16][K] f16 (rows 8..15 zero), W: [C][K] f16 (optionally per-batch).
// Writes rows 0..7 only.  One wave per (b, 16-col tile).
// ---------------------------------------------------------------------------
__global__ __launch_bounds__(32) void gemm16_k(
    const _Float16* __restrict__ A, long long a_bstride,
    const _Float16* __restrict__ W, long long w_bstride,
    const float* __restrict__ bias, const float* __restrict__ resid,
    float* __restrict__ outF, _Float16* __restrict__ outH,
    int K, int C, int act, float postscale) {
  int b = blockIdx.x, ct = blockIdx.y, lane = threadIdx.x;
  const _Float16* Ab = A + (size_t)b * a_bstride;
  const _Float16* Wb = W + (size_t)b * w_bstride;
  v8f acc = {};
  for (int k0 = 0; k0 < K; k0 += 32) {
    v16h a  = load_a16(Ab, K, k0, lane);
    v16h bm = load_b16(Wb, K, ct * 16, k0, lane);
    acc = wmma16(a, bm, acc);
  }
  int col = ct * 16 + (lane & 15);
  int rbase = (lane >> 4) * 8;
  float bv = bias ? bias[col] : 0.f;
#pragma unroll
  for (int r = 0; r < 8; ++r) {
    int row = rbase + r;
    if (row < 8) {
      float val = (acc[r] + bv) * postscale;
      size_t o = ((size_t)b * 16 + row) * C + col;
      if (resid) val += resid[o];
      if (act == 1) val = fmaxf(val, 0.f);
      if (outF) outF[o] = val;
      if (outH) outH[o] = (_Float16)val;
    }
  }
}

// ---------------------------------------------------------------------------
// Attention per batch element: logits = q @ k^T (already SCALE'd q),
// softmax over n with reference's (+1e-8, renormalize), emit f16 attn.
// ---------------------------------------------------------------------------
__global__ __launch_bounds__(256) void attn_k(
    const _Float16* __restrict__ q,     // [B][16][D]
    const _Float16* __restrict__ kmat,  // [B*N][D]
    _Float16* __restrict__ attn) {      // [B][16][N], rows 8..15 pre-zeroed
  __shared__ float logits[8][N_];  // 64KB
  int b = blockIdx.x;
  int wave = threadIdx.x >> 5, lane = threadIdx.x & 31;
  const _Float16* qb = q + (size_t)b * 16 * D_;
  const _Float16* kb = kmat + (size_t)b * N_ * D_;

  v16h afr[8];
#pragma unroll
  for (int ks = 0; ks < 8; ++ks) afr[ks] = load_a16(qb, D_, ks * 32, lane);

  for (int ct = wave; ct < N_ / 16; ct += 8) {
    v8f acc = {};
#pragma unroll
    for (int ks = 0; ks < 8; ++ks) {
      v16h bf = load_b16(kb, D_, ct * 16, ks * 32, lane);
      acc = wmma16(afr[ks], bf, acc);
    }
    if (lane < 16) {
      int col = ct * 16 + lane;
#pragma unroll
      for (int r = 0; r < 8; ++r) logits[r][col] = acc[r];
    }
  }
  __syncthreads();

  // softmax: wave w handles row w
  float* row = logits[wave];
  float m = -3.402823466e38f;
  for (int i = lane; i < N_; i += 32) m = fmaxf(m, row[i]);
#pragma unroll
  for (int off = 16; off; off >>= 1) m = fmaxf(m, __shfl_xor(m, off, 32));
  float s = 0.f;
  for (int i = lane; i < N_; i += 32) s += expf(row[i] - m);
#pragma unroll
  for (int off = 16; off; off >>= 1) s += __shfl_xor(s, off, 32);
  float inv = 1.f / s;
  const float corr = 1.f / (1.f + (float)N_ * 1e-8f);
  _Float16* arow = attn + ((size_t)b * 16 + wave) * N_;
  for (int i = lane; i < N_; i += 32)
    arow[i] = (_Float16)((expf(row[i] - m) * inv + 1e-8f) * corr);
}

// ---------------------------------------------------------------------------
// GRU gate combine (torch order r,z,n):  slots2 = (1-z)*n + z*prev
// ---------------------------------------------------------------------------
__global__ void gru_gate_k(const float* __restrict__ gi, const float* __restrict__ gh,
                           const float* __restrict__ prev, float* __restrict__ slots2) {
  int i = blockIdx.x * blockDim.x + threadIdx.x;  // over B*8*D
  if (i >= B_ * S_ * D_) return;
  int c = i & (D_ - 1);
  int s = (i >> 8) & 7;
  int b = i >> 11;
  size_t gb = ((size_t)b * 16 + s) * (3 * D_);
  float ir = gi[gb + c], iz = gi[gb + D_ + c], in = gi[gb + 2 * D_ + c];
  float hr = gh[gb + c], hz = gh[gb + D_ + c], hn = gh[gb + 2 * D_ + c];
  float r = 1.f / (1.f + expf(-(ir + hr)));
  float z = 1.f / (1.f + expf(-(iz + hz)));
  float n = tanhf(in + r * hn);
  size_t sb = ((size_t)b * 16 + s) * D_ + c;
  slots2[sb] = (1.f - z) * n + z * prev[sb];
}

// ---------------------------------------------------------------------------
// Host orchestration
// ---------------------------------------------------------------------------
extern "C" void kernel_launch(void* const* d_in, const int* in_sizes, int n_in,
                              void* d_out, int out_size, void* d_ws, size_t ws_size,
                              hipStream_t stream) {
  (void)in_sizes; (void)n_in; (void)out_size; (void)ws_size;
  const float* x     = (const float*)d_in[0];
  const float* noise = (const float*)d_in[1];
  const float* mu    = (const float*)d_in[2];
  const float* sigma = (const float*)d_in[3];
  const float* Wq = (const float*)d_in[4];   const float* bq = (const float*)d_in[5];
  const float* Wk = (const float*)d_in[6];   const float* bk = (const float*)d_in[7];
  const float* Wv = (const float*)d_in[8];   const float* bv = (const float*)d_in[9];
  const float* Wih = (const float*)d_in[10]; const float* bih = (const float*)d_in[11];
  const float* Whh = (const float*)d_in[12]; const float* bhh = (const float*)d_in[13];
  const float* W1 = (const float*)d_in[14];  const float* b1 = (const float*)d_in[15];
  const float* W2 = (const float*)d_in[16];  const float* b2 = (const float*)d_in[17];
  const float* g_in = (const float*)d_in[18]; const float* be_in = (const float*)d_in[19];
  const float* g_sl = (const float*)d_in[20]; const float* be_sl = (const float*)d_in[21];
  const float* g_ff = (const float*)d_in[22]; const float* be_ff = (const float*)d_in[23];

  char* ws = (char*)d_ws;
  size_t off = 0;
  auto alloc = [&](size_t bytes) -> char* {
    char* p = ws + off;
    off = (off + bytes + 255) & ~(size_t)255;
    return p;
  };

  _Float16* k_h  = (_Float16*)alloc((size_t)B_ * N_ * D_ * 2);
  _Float16* vT_h = (_Float16*)alloc((size_t)B_ * D_ * N_ * 2);
  _Float16* wq_h  = (_Float16*)alloc((size_t)D_ * D_ * 2);
  _Float16* wk_h  = (_Float16*)alloc((size_t)D_ * D_ * 2);
  _Float16* wv_h  = (_Float16*)alloc((size_t)D_ * D_ * 2);
  _Float16* wih_h = (_Float16*)alloc((size_t)3 * D_ * D_ * 2);
  _Float16* whh_h = (_Float16*)alloc((size_t)3 * D_ * D_ * 2);
  _Float16* w1_h  = (_Float16*)alloc((size_t)D_ * D_ * 2);
  _Float16* w2_h  = (_Float16*)alloc((size_t)D_ * D_ * 2);

  size_t zstart = off;  // everything below is zero-filled each launch
  float* slots   = (float*)alloc((size_t)B_ * 16 * D_ * 4);
  float* slots2  = (float*)alloc((size_t)B_ * 16 * D_ * 4);
  float* gi      = (float*)alloc((size_t)B_ * 16 * 3 * D_ * 4);
  float* gh      = (float*)alloc((size_t)B_ * 16 * 3 * D_ * 4);
  _Float16* slots_h   = (_Float16*)alloc((size_t)B_ * 16 * D_ * 2);
  _Float16* slotsln_h = (_Float16*)alloc((size_t)B_ * 16 * D_ * 2);
  _Float16* q_h       = (_Float16*)alloc((size_t)B_ * 16 * D_ * 2);
  _Float16* updates_h = (_Float16*)alloc((size_t)B_ * 16 * D_ * 2);
  _Float16* ffln_h    = (_Float16*)alloc((size_t)B_ * 16 * D_ * 2);
  _Float16* h_h       = (_Float16*)alloc((size_t)B_ * 16 * D_ * 2);
  _Float16* attn_h    = (_Float16*)alloc((size_t)B_ * 16 * N_ * 2);
  size_t zbytes = off - zstart;

  // Zero the padded small buffers (rows 8..15 of all 16-row tiles must be 0)
  {
    size_t n4 = zbytes / 16;
    zero_f4_k<<<(unsigned)((n4 + 255) / 256), 256, 0, stream>>>((float4*)(ws + zstart), n4);
  }
  // Weight conversions to f16
  cvt_h_k<<<(D_ * D_ + 255) / 256, 256, 0, stream>>>(Wq, wq_h, D_ * D_);
  cvt_h_k<<<(D_ * D_ + 255) / 256, 256, 0, stream>>>(Wk, wk_h, D_ * D_);
  cvt_h_k<<<(D_ * D_ + 255) / 256, 256, 0, stream>>>(Wv, wv_h, D_ * D_);
  cvt_h_k<<<(3 * D_ * D_ + 255) / 256, 256, 0, stream>>>(Wih, wih_h, 3 * D_ * D_);
  cvt_h_k<<<(3 * D_ * D_ + 255) / 256, 256, 0, stream>>>(Whh, whh_h, 3 * D_ * D_);
  cvt_h_k<<<(D_ * D_ + 255) / 256, 256, 0, stream>>>(W1, w1_h, D_ * D_);
  cvt_h_k<<<(D_ * D_ + 255) / 256, 256, 0, stream>>>(W2, w2_h, D_ * D_);

  // Slot init
  slot_init_k<<<(B_ * S_ * D_ + 255) / 256, 256, 0, stream>>>(mu, sigma, noise, slots);

  // Fused LN(x) + K/V projections
  lnx_kv_k<<<B_ * N_ / 16, 256, 0, stream>>>(x, g_in, be_in, wk_h, bk, wv_h, bv, k_h, vT_h);

  for (int it = 0; it < ITERS_; ++it) {
    // LN(slots) with slot norm params; also plain f16 copy of prev slots
    ln16_k<<<64, 256, 0, stream>>>(slots, g_sl, be_sl, slotsln_h, slots_h);
    // q = (LN(slots) @ Wq^T + bq) * SCALE
    gemm16_k<<<dim3(B_, D_ / 16), 32, 0, stream>>>(
        slotsln_h, (long long)16 * D_, wq_h, 0, bq, nullptr, nullptr, q_h,
        D_, D_, 0, SCALE_);
    // attention: logits, softmax(+1e-8 renorm)
    attn_k<<<B_, 256, 0, stream>>>(q_h, k_h, attn_h);
    // updates = attn @ v  (v stored transposed per batch)
    gemm16_k<<<dim3(B_, D_ / 16), 32, 0, stream>>>(
        attn_h, (long long)16 * N_, vT_h, (long long)D_ * N_, nullptr, nullptr,
        nullptr, updates_h, N_, D_, 0, 1.0f);
    // GRU gates
    gemm16_k<<<dim3(B_, 3 * D_ / 16), 32, 0, stream>>>(
        updates_h, (long long)16 * D_, wih_h, 0, bih, nullptr, gi, nullptr,
        D_, 3 * D_, 0, 1.0f);
    gemm16_k<<<dim3(B_, 3 * D_ / 16), 32, 0, stream>>>(
        slots_h, (long long)16 * D_, whh_h, 0, bhh, nullptr, gh, nullptr,
        D_, 3 * D_, 0, 1.0f);
    gru_gate_k<<<(B_ * S_ * D_ + 255) / 256, 256, 0, stream>>>(gi, gh, slots, slots2);
    // residual MLP
    ln16_k<<<64, 256, 0, stream>>>(slots2, g_ff, be_ff, ffln_h, nullptr);
    gemm16_k<<<dim3(B_, D_ / 16), 32, 0, stream>>>(
        ffln_h, (long long)16 * D_, w1_h, 0, b1, nullptr, nullptr, h_h,
        D_, D_, 1, 1.0f);
    gemm16_k<<<dim3(B_, D_ / 16), 32, 0, stream>>>(
        h_h, (long long)16 * D_, w2_h, 0, b2, slots2, slots, nullptr,
        D_, D_, 0, 1.0f);
  }

  copy_out_k<<<(B_ * S_ * D_ + 255) / 256, 256, 0, stream>>>(slots, (float*)d_out);
}
